// SpatialGNN_26225070310038
// MI455X (gfx1250) — matrix operand order repaired and checked by reading, not compiled, not run on hardware
//
#include <hip/hip_runtime.h>
#include <hip/hip_bf16.h>

typedef _Float16 half_t;
typedef __attribute__((ext_vector_type(16))) _Float16     v16h;
typedef __attribute__((ext_vector_type(8)))  float        v8f;
typedef __attribute__((ext_vector_type(8)))  unsigned int v8u;

union FragU { v8u u; v16h h; };

__device__ inline unsigned int pack2(float a, float b) {
    union { _Float16 h[2]; unsigned int u; } p;
    p.h[0] = (_Float16)a; p.h[1] = (_Float16)b;
    return p.u;
}

// low 32 bits of a generic pointer to LDS == workgroup-relative LDS byte address
__device__ inline unsigned lds_off(const void* p) {
    return (unsigned)(unsigned long long)p;
}

// async DMA: 16 bytes/lane from global (saddr + 32-bit voffset) into LDS
__device__ inline void async_copy_b128(unsigned ldsAddr, unsigned gOffset,
                                       const void* base) {
    asm volatile("global_load_async_to_lds_b128 %0, %1, %2"
                 :: "v"(ldsAddr), "v"(gOffset), "s"(base)
                 : "memory");
}

// ---------------------------------------------------------------------------
// f32 -> f16 weight conversion (grid-stride)
// ---------------------------------------------------------------------------
__global__ void cvt_f16_kernel(const float* __restrict__ src,
                               half_t* __restrict__ dst, int n) {
    int i = blockIdx.x * blockDim.x + threadIdx.x;
    int stride = gridDim.x * blockDim.x;
    for (; i < n; i += stride) dst[i] = (half_t)src[i];
}

// ---------------------------------------------------------------------------
// Fused GCNConv: one block per graph (32 nodes, 256 edges).
//   h = x @ Wc^T (WMMA, 2x4 tiles over 8 waves), degree + rsqrt in LDS,
//   edge scatter-add in LDS (ds_add_f32), self-loop + bias + relu,
//   pack f16 into Hin[B, 2048].
// ---------------------------------------------------------------------------
__global__ __launch_bounds__(256)
void gcn_fused_kernel(const float* __restrict__ x,
                      const unsigned int* __restrict__ WcU,  // Wc f16 pairs [64][32]
                      const float* __restrict__ bc,
                      const int* __restrict__ esrc,
                      const int* __restrict__ edst,
                      half_t* __restrict__ Hin) {
    __shared__ unsigned int xl[32 * 34];   // 32 rows x 32 f16-pairs (+2 pad)
    __shared__ float hl[32 * 72];          // post-linear h
    __shared__ float accl[32 * 72];        // scatter accumulator
    __shared__ float degl[32];
    __shared__ float dinvl[32];

    const int g    = blockIdx.x;
    const int t    = threadIdx.x;
    const int lane = t & 31;
    const int wave = t >> 5;
    const int sel  = lane >> 4;     // half-wave select
    const int r_   = lane & 15;

    // stage x tile (32x64 f32) as f16 pairs
    #pragma unroll
    for (int i = 0; i < 4; ++i) {
        int q = t * 4 + i;                 // 1024 pairs
        int row = q >> 5, pc = q & 31;
        const float* xp = x + (size_t)(g * 32 + row) * 64 + pc * 2;
        xl[row * 34 + pc] = pack2(xp[0], xp[1]);
    }
    if (t < 32) degl[t] = 1.0f;            // self-loop
    __syncthreads();

    // h = x @ Wc^T : wave -> (mb in 0..1, nb in 0..3) 16x16 tile
    {
        const int mb = wave >> 2, nb = wave & 3;
        v8f c = {};
        #pragma unroll
        for (int kb = 0; kb < 64; kb += 32) {
            FragU a, b;
            #pragma unroll
            for (int v = 0; v < 8; ++v) {
                int p = v + sel * 8;
                a.u[v] = xl[(mb * 16 + r_) * 34 + (kb >> 1) + p];
                b.u[v] = WcU[(nb * 16 + r_) * 32 + (kb >> 1) + sel * 8 + v];
            }
            c = __builtin_amdgcn_wmma_f32_16x16x32_f16(
                    false, a.h, false, b.h, (short)0, c, false, false);
        }
        #pragma unroll
        for (int v = 0; v < 8; ++v)
            hl[(mb * 16 + v + sel * 8) * 72 + nb * 16 + r_] = c[v];
    }

    // zero accumulator
    #pragma unroll
    for (int i = 0; i < 8; ++i) {
        int idx = t + 256 * i;
        accl[(idx >> 6) * 72 + (idx & 63)] = 0.0f;
    }
    __syncthreads();

    // in-degree (incl. self-loop already seeded)
    const int e = g * 256 + t;
    const int s = esrc[e] & 31;
    const int d = edst[e] & 31;
    atomicAdd(&degl[d], 1.0f);
    __syncthreads();
    if (t < 32) dinvl[t] = rsqrtf(degl[t]);
    __syncthreads();

    // message scatter: acc[d] += h[s] * dinv[s]*dinv[d]
    {
        float norm = dinvl[s] * dinvl[d];
        #pragma unroll 8
        for (int f = 0; f < 64; ++f)
            atomicAdd(&accl[d * 72 + f], hl[s * 72 + f] * norm);
    }
    __syncthreads();

    // self-loop term + bias + relu, pack f16 into Hin row
    half_t* out = Hin + (size_t)g * 2048;
    #pragma unroll
    for (int i = 0; i < 8; ++i) {
        int idx = t + 256 * i;
        int n = idx >> 6, f = idx & 63;
        float di = dinvl[n];
        float v = accl[n * 72 + f] + hl[n * 72 + f] * di * di + bc[f];
        v = v > 0.0f ? v : 0.0f;
        out[idx] = (half_t)v;              // column = n*64 + f == idx
    }
}

// ---------------------------------------------------------------------------
// lin1: H2 = relu(Hin[16384,2048] @ W1^T + b1), f16 in / f32 acc / f16 out.
// Block tile 128(M) x 128(N), 8 waves = 4(M) x 2(N), wave tile 32x64.
// A/B tiles double-buffered in LDS via global_load_async_to_lds_b128,
// pipelined with s_wait_asynccnt.
// LDS row stride = 20 dwords (80B: 16B-aligned for B128, conflict-free frags).
// ---------------------------------------------------------------------------
__global__ __launch_bounds__(256)
void lin1_kernel(const unsigned int* __restrict__ A,   // Hin pairs [16384][1024]
                 const unsigned int* __restrict__ Bw,  // W1  pairs [2048][1024]
                 const float* __restrict__ b1,
                 half_t* __restrict__ H2) {
    __shared__ unsigned int aLds[2][128 * 20];
    __shared__ unsigned int bLds[2][128 * 20];

    const int t    = threadIdx.x;
    const int lane = t & 31;
    const int wave = t >> 5;
    const int sel  = lane >> 4;
    const int r_   = lane & 15;
    const int wm   = wave & 3;                // 0..3 over M
    const int wn   = wave >> 2;               // 0..1 over N
    const int m0   = blockIdx.x * 128;
    const int n0   = blockIdx.y * 128;

    // async-copy chunk mapping: 512 x 16B chunks per tile, 2 per thread
    unsigned aGO[2], bGO[2];
    unsigned aLA[2][2], bLA[2][2];            // [buf][chunk] LDS byte addrs
    #pragma unroll
    for (int i = 0; i < 2; ++i) {
        unsigned q = (unsigned)t + 256u * i;
        unsigned row = q >> 2, c4 = q & 3;
        aGO[i] = (unsigned)(m0 + row) * 4096u + c4 * 16u;   // + kbyte later
        bGO[i] = (unsigned)(n0 + row) * 4096u + c4 * 16u;
        unsigned rel = row * 80u + c4 * 16u;
        aLA[0][i] = lds_off(&aLds[0][0]) + rel;
        aLA[1][i] = lds_off(&aLds[1][0]) + rel;
        bLA[0][i] = lds_off(&bLds[0][0]) + rel;
        bLA[1][i] = lds_off(&bLds[1][0]) + rel;
    }

    auto issue = [&](int buf, unsigned kbyte) {
        #pragma unroll
        for (int i = 0; i < 2; ++i) {
            async_copy_b128(aLA[buf][i], aGO[i] + kbyte, A);
            async_copy_b128(bLA[buf][i], bGO[i] + kbyte, Bw);
        }
    };

    v8f c[2][4] = {};

    issue(0, 0);                              // prologue: tile 0 -> buf 0
    for (int kk = 0; kk < 64; ++kk) {         // 64 K-steps of 32 (16 pairs)
        const int cur = kk & 1;
        if (kk + 1 < 64) {
            issue(cur ^ 1, (unsigned)(kk + 1) * 64u);      // 16 pairs * 4B
            asm volatile("s_wait_asynccnt 0x4" ::: "memory"); // tile kk done
        } else {
            asm volatile("s_wait_asynccnt 0x0" ::: "memory");
        }
        __syncthreads();                      // tile kk visible to all waves

        const unsigned int* aB = aLds[cur];
        const unsigned int* bB = bLds[cur];

        FragU bf[4];
        #pragma unroll
        for (int nb = 0; nb < 4; ++nb)
            #pragma unroll
            for (int v = 0; v < 8; ++v)
                bf[nb].u[v] = bB[(wn * 64 + nb * 16 + r_) * 20 + sel * 8 + v];

        #pragma unroll
        for (int mb = 0; mb < 2; ++mb) {
            FragU a;
            #pragma unroll
            for (int v = 0; v < 8; ++v) {
                int p = v + sel * 8;
                a.u[v] = aB[(wm * 32 + mb * 16 + r_) * 20 + p];
            }
            #pragma unroll
            for (int nb = 0; nb < 4; ++nb)
                c[mb][nb] = __builtin_amdgcn_wmma_f32_16x16x32_f16(
                    false, a.h, false, bf[nb].h, (short)0, c[mb][nb], false, false);
        }
        __syncthreads();                      // all waves done with buf[cur]
    }

    // epilogue: bias + relu + f16 pack
    #pragma unroll
    for (int mb = 0; mb < 2; ++mb)
        #pragma unroll
        for (int nb = 0; nb < 4; ++nb) {
            int n = n0 + wn * 64 + nb * 16 + r_;
            float bias = b1[n];
            #pragma unroll
            for (int v = 0; v < 8; ++v) {
                int m = m0 + wm * 32 + mb * 16 + v + sel * 8;
                float val = c[mb][nb][v] + bias;
                val = val > 0.0f ? val : 0.0f;
                H2[(size_t)m * 2048 + n] = (half_t)val;
            }
        }
}

// ---------------------------------------------------------------------------
// lin2 + softmax: one wave per batch row, shuffle reduction.
// ---------------------------------------------------------------------------
__global__ __launch_bounds__(128)
void lin2_softmax_kernel(const unsigned int* __restrict__ H2U,  // pairs [16384][1024]
                         const float* __restrict__ W2,          // [2][2048]
                         const float* __restrict__ b2,
                         float* __restrict__ out) {
    const int wave = threadIdx.x >> 5;
    const int lane = threadIdx.x & 31;
    const int b = blockIdx.x * 4 + wave;
    const unsigned int* h = H2U + (size_t)b * 1024;

    float a0 = 0.0f, a1 = 0.0f;
    for (int i = lane; i < 1024; i += 32) {
        union { unsigned int u; _Float16 hh[2]; } cv; cv.u = h[i];
        float x0 = (float)cv.hh[0], x1 = (float)cv.hh[1];
        int k = i * 2;
        a0 += x0 * W2[k]        + x1 * W2[k + 1];
        a1 += x0 * W2[2048 + k] + x1 * W2[2048 + k + 1];
    }
    #pragma unroll
    for (int off = 16; off > 0; off >>= 1) {
        a0 += __shfl_down(a0, off, 32);
        a1 += __shfl_down(a1, off, 32);
    }
    if (lane == 0) {
        float z0 = a0 + b2[0], z1 = a1 + b2[1];
        float m = fmaxf(z0, z1);
        float e0 = __expf(z0 - m), e1 = __expf(z1 - m);
        float s = e0 + e1;
        out[2 * b]     = e0 / s;
        out[2 * b + 1] = e1 / s;
    }
}

// ---------------------------------------------------------------------------
extern "C" void kernel_launch(void* const* d_in, const int* in_sizes, int n_in,
                              void* d_out, int out_size, void* d_ws, size_t ws_size,
                              hipStream_t stream) {
    (void)n_in; (void)out_size; (void)ws_size;
    const float* x  = (const float*)d_in[0];
    const float* Wc = (const float*)d_in[1];
    const float* bc = (const float*)d_in[2];
    const float* W1 = (const float*)d_in[3];
    const float* b1 = (const float*)d_in[4];
    const float* W2 = (const float*)d_in[5];
    const float* b2 = (const float*)d_in[6];
    const int*   ei = (const int*)d_in[7];    // [2, E] flat: src then dst
    // d_in[8] (batch) unused: graph id == node/32

    char* ws = (char*)d_ws;
    half_t* WcH  = (half_t*)(ws);                                   // 8 KB
    half_t* W1H  = (half_t*)(ws + 8192);                            // 8 MB
    half_t* HinH = (half_t*)(ws + 8192 + 8388608);                  // 64 MB
    half_t* H2H  = (half_t*)(ws + 8192 + 8388608 + 67108864);       // 64 MB

    const int E = in_sizes[7] / 2;

    cvt_f16_kernel<<<64, 256, 0, stream>>>(Wc, WcH, 64 * 64);
    cvt_f16_kernel<<<4096, 256, 0, stream>>>(W1, W1H, 2048 * 2048);

    gcn_fused_kernel<<<16384, 256, 0, stream>>>(
        x, (const unsigned int*)WcH, bc, ei, ei + E, HinH);

    dim3 g1(16384 / 128, 2048 / 128);
    lin1_kernel<<<g1, 256, 0, stream>>>(
        (const unsigned int*)HinH, (const unsigned int*)W1H, b1, H2H);

    lin2_softmax_kernel<<<16384 / 4, 128, 0, stream>>>(
        (const unsigned int*)H2H, W2, b2, (float*)d_out);
}
